// TemporalGNN_39676907880506
// MI455X (gfx1250) — matrix operand order repaired and checked by reading, not compiled, not run on hardware
//
#include <hip/hip_runtime.h>

typedef __bf16 bf16;
typedef __attribute__((ext_vector_type(16))) __bf16 v16bf;
typedef __attribute__((ext_vector_type(8)))  __bf16 v8bf;
typedef __attribute__((ext_vector_type(8)))  float  v8f;

#define NND   10000
#define TT    32
#define NFEAT 64
#define HID   128

static __device__ __forceinline__ float sigf(float x){ return 1.0f/(1.0f+expf(-x)); }

// A fragment (16x32 bf16, lane owns row M = lane&15): two v8bf chunks per ISA layout
static __device__ __forceinline__ v16bf load_afrag(const bf16* arow, int hi){
  union { v16bf v; v8bf h[2]; } a;
  a.h[0] = *(const v8bf*)(arow + hi*8);        // elems 0..7 : K = hi*8 .. +7
  a.h[1] = *(const v8bf*)(arow + 16 + hi*8);   // elems 8..15: K = 16+hi*8 .. +7
  return a.v;
}
// B fragment (32x16) from row-major (N,K) weights: two v8bf chunks, same K order as A
static __device__ __forceinline__ v16bf load_bfrag(const bf16* wrow, int hi){
  union { v16bf v; v8bf h[2]; } b;
  b.h[0] = *(const v8bf*)(wrow + hi*8);
  b.h[1] = *(const v8bf*)(wrow + 16 + hi*8);
  return b.v;
}
static __device__ __forceinline__ v8f wmma_acc(v16bf a, v16bf b, v8f c){
  return __builtin_amdgcn_wmma_f32_16x16x32_bf16(false, a, false, b, (short)0, c, false, false);
}

// ---------------- generic WMMA GEMM: D = act(A @ W^T + bias) ----------------
// Each wave computes a 16(M) x 64(N) strip: A fragment loaded once per K-step,
// 4 back-to-back WMMAs for 4x arithmetic intensity.  N % 64 == 0 required.
// gather mode (gsrc!=null): K==256, row = concat(Ah[gsrc[r]], Ah[gdst[r]]), Ah rows of 128.
__global__ void tg_gemm(const bf16* __restrict__ A, long lda, long aoff,
                        const bf16* __restrict__ W, int K,
                        const float* __restrict__ bias,
                        float* __restrict__ Df, bf16* __restrict__ Dbf,
                        int M, int N, int act,
                        const int* __restrict__ gsrc, const int* __restrict__ gdst){
  int wid = blockIdx.x * (blockDim.x >> 5) + (threadIdx.x >> 5);
  int NG = N >> 6;                                  // 64-wide N groups
  if (wid >= (M >> 4) * NG) return;
  int mt = wid / NG, ng = wid - mt * NG;
  int lane = threadIdx.x & 31, hi = lane >> 4, ln = lane & 15;
  int row = mt*16 + ln;
  const bf16* wb0 = W + (long)(ng*64      + ln) * K;
  const bf16* wb1 = W + (long)(ng*64 + 16 + ln) * K;
  const bf16* wb2 = W + (long)(ng*64 + 32 + ln) * K;
  const bf16* wb3 = W + (long)(ng*64 + 48 + ln) * K;
  v8f acc0 = {0.f,0.f,0.f,0.f,0.f,0.f,0.f,0.f};
  v8f acc1 = acc0, acc2 = acc0, acc3 = acc0;
  if (gsrc){
    const bf16* a1 = A + (long)gsrc[row] * HID;
    const bf16* a2 = A + (long)gdst[row] * HID;
    #pragma unroll
    for (int k = 0; k < 2*HID; k += 32){
      v16bf a = load_afrag((k < HID ? a1 + k : a2 + (k - HID)), hi);
      acc0 = wmma_acc(a, load_bfrag(wb0 + k, hi), acc0);
      acc1 = wmma_acc(a, load_bfrag(wb1 + k, hi), acc1);
      acc2 = wmma_acc(a, load_bfrag(wb2 + k, hi), acc2);
      acc3 = wmma_acc(a, load_bfrag(wb3 + k, hi), acc3);
    }
  } else {
    const bf16* arow = A + (long)row * lda + aoff;
    for (int k = 0; k < K; k += 32){
      __builtin_prefetch(arow + k + 64, 0, 1);
      v16bf a = load_afrag(arow + k, hi);
      acc0 = wmma_acc(a, load_bfrag(wb0 + k, hi), acc0);
      acc1 = wmma_acc(a, load_bfrag(wb1 + k, hi), acc1);
      acc2 = wmma_acc(a, load_bfrag(wb2 + k, hi), acc2);
      acc3 = wmma_acc(a, load_bfrag(wb3 + k, hi), acc3);
    }
  }
  v8f* accs[4] = {&acc0, &acc1, &acc2, &acc3};
  #pragma unroll
  for (int q = 0; q < 4; ++q){
    int col = ng*64 + q*16 + ln;
    float bv = bias ? bias[col] : 0.f;
    #pragma unroll
    for (int j = 0; j < 8; ++j){
      int m = mt*16 + j + hi*8;
      float v = (*accs[q])[j] + bv;
      if (act == 1) v = tanhf(v);
      long idx = (long)m * N + col;
      if (Df)  Df[idx]  = v;
      if (Dbf) Dbf[idx] = (bf16)v;
    }
  }
}

// ---------------- LSTM gate GEMM: G = x_t @ Wih^T + h @ Whh^T + b  (N=512) --
// Wave computes 16x64 strip of the 512-wide gate matrix.
__global__ void tg_lstm_gate(const bf16* __restrict__ xseq, int in_d, int t,
                             const bf16* __restrict__ hbf,
                             const bf16* __restrict__ Wih, const bf16* __restrict__ Whh,
                             const float* __restrict__ bias, float* __restrict__ G, int M){
  int wid = blockIdx.x * (blockDim.x >> 5) + (threadIdx.x >> 5);
  const int NG = 512 >> 6;
  if (wid >= (M >> 4) * NG) return;
  int mt = wid / NG, ng = wid - mt * NG;
  int lane = threadIdx.x & 31, hi = lane >> 4, ln = lane & 15;
  int row = mt*16 + ln;
  const bf16* ax = xseq + ((long)row * TT + t) * in_d;
  const bf16* ah = hbf  + (long)row * HID;
  const bf16* wi0 = Wih + (long)(ng*64      + ln) * in_d;
  const bf16* wi1 = Wih + (long)(ng*64 + 16 + ln) * in_d;
  const bf16* wi2 = Wih + (long)(ng*64 + 32 + ln) * in_d;
  const bf16* wi3 = Wih + (long)(ng*64 + 48 + ln) * in_d;
  const bf16* wh0 = Whh + (long)(ng*64      + ln) * HID;
  const bf16* wh1 = Whh + (long)(ng*64 + 16 + ln) * HID;
  const bf16* wh2 = Whh + (long)(ng*64 + 32 + ln) * HID;
  const bf16* wh3 = Whh + (long)(ng*64 + 48 + ln) * HID;
  v8f acc0 = {0.f,0.f,0.f,0.f,0.f,0.f,0.f,0.f};
  v8f acc1 = acc0, acc2 = acc0, acc3 = acc0;
  for (int k = 0; k < in_d; k += 32){
    v16bf a = load_afrag(ax + k, hi);
    acc0 = wmma_acc(a, load_bfrag(wi0 + k, hi), acc0);
    acc1 = wmma_acc(a, load_bfrag(wi1 + k, hi), acc1);
    acc2 = wmma_acc(a, load_bfrag(wi2 + k, hi), acc2);
    acc3 = wmma_acc(a, load_bfrag(wi3 + k, hi), acc3);
  }
  #pragma unroll
  for (int k = 0; k < HID; k += 32){
    v16bf a = load_afrag(ah + k, hi);
    acc0 = wmma_acc(a, load_bfrag(wh0 + k, hi), acc0);
    acc1 = wmma_acc(a, load_bfrag(wh1 + k, hi), acc1);
    acc2 = wmma_acc(a, load_bfrag(wh2 + k, hi), acc2);
    acc3 = wmma_acc(a, load_bfrag(wh3 + k, hi), acc3);
  }
  v8f* accs[4] = {&acc0, &acc1, &acc2, &acc3};
  #pragma unroll
  for (int q = 0; q < 4; ++q){
    int col = ng*64 + q*16 + ln;
    float bv = bias[col];
    #pragma unroll
    for (int j = 0; j < 8; ++j)
      G[(long)(mt*16 + j + hi*8) * 512 + col] = (*accs[q])[j] + bv;
  }
}

// ---------------- LSTM cell elementwise ------------------------------------
__global__ void tg_lstm_cell(const float* __restrict__ G, float* __restrict__ c,
                             bf16* __restrict__ hbf, bf16* __restrict__ seqo, int t, int M){
  long i = (long)blockIdx.x * blockDim.x + threadIdx.x;
  if (i >= (long)M * HID) return;
  int row = (int)(i >> 7), j = (int)(i & 127);
  long gb = (long)row * 512 + j;
  float gi = G[gb], gf = G[gb+128], gg = G[gb+256], go = G[gb+384];
  float cn = sigf(gf) * c[i] + sigf(gi) * tanhf(gg);
  float h  = sigf(go) * tanhf(cn);
  c[i] = cn; hbf[i] = (bf16)h;
  seqo[((long)row * TT + t) * 256 + j] = (bf16)h;   // caller pre-offsets by dir*128
}

// ---------------- elementwise / conversion helpers -------------------------
__global__ void tg_f2bf(const float* __restrict__ a, bf16* __restrict__ o, long n){
  long i = (long)blockIdx.x * blockDim.x + threadIdx.x; if (i < n) o[i] = (bf16)a[i];
}
__global__ void tg_vadd(const float* a, const float* b, float* o, int n){
  int i = blockIdx.x * blockDim.x + threadIdx.x; if (i < n) o[i] = a[i] + b[i];
}
__global__ void tg_copyf(const float* a, float* o, int n){
  int i = blockIdx.x * blockDim.x + threadIdx.x; if (i < n) o[i] = a[i];
}
__global__ void tg_xcvt(const float* __restrict__ x, const float* __restrict__ fw,
                        bf16* __restrict__ o, long n){
  long i = (long)blockIdx.x * blockDim.x + threadIdx.x;
  if (i < n) o[i] = (bf16)(x[i] * fw[i & (NFEAT-1)]);
}

// ---------------- LayerNorm(h + hnew + gatbias) ----------------------------
__global__ void tg_ln(const float* __restrict__ h, const float* __restrict__ hn,
                      const float* __restrict__ gbias, const float* __restrict__ g,
                      const float* __restrict__ b, float* __restrict__ ho,
                      bf16* __restrict__ hbf, bf16* __restrict__ jk, int jkoff){
  int row = blockIdx.x, c = threadIdx.x;       // blockDim = 128
  __shared__ float s1[HID], s2[HID];
  long i = (long)row * HID + c;
  float z = h[i] + hn[i] + gbias[c];
  s1[c] = z; s2[c] = z*z; __syncthreads();
  for (int off = 64; off; off >>= 1){
    if (c < off){ s1[c] += s1[c+off]; s2[c] += s2[c+off]; } __syncthreads();
  }
  float m = s1[0] * (1.f/HID), v = s2[0] * (1.f/HID) - m*m;
  float y = (z - m) * rsqrtf(v + 1e-5f) * g[c] + b[c];
  ho[i] = y; hbf[i] = (bf16)y; jk[(long)row * 256 + jkoff + c] = (bf16)y;
}

// ---------------- BatchNorm column stats + apply ---------------------------
__global__ void tg_colstats(const float* __restrict__ X, int M, int N,
                            float* __restrict__ mean, float* __restrict__ istd){
  int col = blockIdx.x, tid = threadIdx.x;     // blockDim = 256
  float s = 0.f, q = 0.f;
  for (int r = tid; r < M; r += 256){ float v = X[(long)r*N + col]; s += v; q += v*v; }
  __shared__ float a1[256], a2[256];
  a1[tid] = s; a2[tid] = q; __syncthreads();
  for (int off = 128; off; off >>= 1){
    if (tid < off){ a1[tid] += a1[tid+off]; a2[tid] += a2[tid+off]; } __syncthreads();
  }
  if (tid == 0){
    float m = a1[0]/M, v = a2[0]/M - m*m;
    mean[col] = m; istd[col] = rsqrtf(v + 1e-5f);
  }
}
__global__ void tg_bnrelu(const float* __restrict__ X, const float* mean, const float* istd,
                          const float* g, const float* b, float* __restrict__ Yf,
                          bf16* __restrict__ Ybf, long n, int N){
  long i = (long)blockIdx.x * blockDim.x + threadIdx.x; if (i >= n) return;
  int c = (int)(i % N);
  float y = (X[i] - mean[c]) * istd[c] * g[c] + b[c];
  y = y > 0.f ? y : 0.f;
  if (Yf) Yf[i] = y; if (Ybf) Ybf[i] = (bf16)y;
}

// ---------------- GATv2 attention pieces -----------------------------------
static __device__ __forceinline__ unsigned flipf(float f){
  unsigned u = __float_as_uint(f); return (u & 0x80000000u) ? ~u : (u | 0x80000000u);
}
static __device__ __forceinline__ float unflipf(unsigned u){
  unsigned b = (u & 0x80000000u) ? (u ^ 0x80000000u) : ~u; return __uint_as_float(b);
}
__global__ void tg_alpha(const float* __restrict__ xl, const float* __restrict__ xr,
                         const float* __restrict__ se, const float* __restrict__ We,
                         const float* __restrict__ att, const int* __restrict__ src,
                         const int* __restrict__ dst, float* __restrict__ alpha,
                         unsigned* __restrict__ amax, int E){
  int e = blockIdx.x * blockDim.x + threadIdx.x; if (e >= E) return;
  int s = src[e], d = dst[e]; float sv = se[e];
  #pragma unroll
  for (int hd = 0; hd < 4; ++hd){
    float acc = 0.f;
    for (int c = 0; c < 32; ++c){
      int k = hd*32 + c;
      float m = xl[(long)s*HID + k] + xr[(long)d*HID + k] + sv * We[k];
      m = m > 0.f ? m : 0.2f * m;
      acc += m * att[k];
    }
    alpha[(long)e*4 + hd] = acc;
    atomicMax(&amax[d*4 + hd], flipf(acc));
  }
}
__global__ void tg_expden(const float* __restrict__ alpha, const unsigned* __restrict__ amax,
                          const int* __restrict__ dst, float* __restrict__ ex,
                          float* __restrict__ den, int E){
  long i = (long)blockIdx.x * blockDim.x + threadIdx.x; if (i >= (long)E*4) return;
  int e = (int)(i >> 2), hd = (int)(i & 3), d = dst[e];
  float v = expf(alpha[i] - unflipf(amax[d*4 + hd]));
  ex[i] = v; atomicAdd(&den[d*4 + hd], v);
}
__global__ void tg_scatter(const float* __restrict__ ex, const float* __restrict__ den,
                           const float* __restrict__ xl, const int* __restrict__ src,
                           const int* __restrict__ dst, float* __restrict__ out, int E){
  int e = blockIdx.x * blockDim.x + threadIdx.x; if (e >= E) return;
  int s = src[e], d = dst[e];
  #pragma unroll
  for (int hd = 0; hd < 4; ++hd){
    float a = ex[(long)e*4 + hd] / den[d*4 + hd];
    for (int c = 0; c < 32; ++c){
      int k = hd*32 + c;
      atomicAdd(&out[(long)d*HID + k], a * xl[(long)s*HID + k]);
    }
  }
}

// ---------------- K-reduce dot (N=1 heads of edge-net / predictor) ---------
__global__ void tg_dot(const bf16* __restrict__ Z, int K, const float* __restrict__ w,
                       const float* __restrict__ b0, float* __restrict__ out, int M, int dotanh){
  int r = blockIdx.x * blockDim.x + threadIdx.x; if (r >= M) return;
  float s = b0[0];
  for (int k = 0; k < K; ++k) s += (float)Z[(long)r*K + k] * w[k];
  out[r] = dotanh ? tanhf(s) : s;
}

// ======================== host side ========================================
static inline int gblk(long waves){ return (int)((waves + 7) / 8); }   // 8 waves / 256-thr block

extern "C" void kernel_launch(void* const* d_in, const int* in_sizes, int n_in,
                              void* d_out, int out_size, void* d_ws, size_t ws_size,
                              hipStream_t stream){
  (void)n_in; (void)out_size; (void)ws_size;
  const int E = in_sizes[1] / 2;
  const float* x   = (const float*)d_in[0];
  const int*   src = (const int*)d_in[1];
  const int*   dst = src + E;
  const float* fw  = (const float*)d_in[2];

  // bump allocator over d_ws
  char* wsp = (char*)d_ws; size_t off = 0;
  auto alloc = [&](size_t bytes)->void*{ off = (off + 255) & ~(size_t)255; void* p = wsp + off; off += bytes; return p; };

  // ---- bf16 weight pack ----
  bf16 *WihB[4], *WhhB[4]; float* Bsum[4];
  const int lstm_base[4] = {3, 7, 11, 15};           // l1f, l1b, l2f, l2b
  const int lstm_ind[4]  = {NFEAT, NFEAT, 256, 256};
  for (int d = 0; d < 4; ++d){
    WihB[d] = (bf16*)alloc((size_t)512 * lstm_ind[d] * 2);
    WhhB[d] = (bf16*)alloc((size_t)512 * HID * 2);
    Bsum[d] = (float*)alloc(512 * 4);
  }
  bf16* projWb = (bf16*)alloc(HID*256*2);
  bf16* enW1b  = (bf16*)alloc(HID*256*2);
  bf16* enW2b  = (bf16*)alloc(64*HID*2);
  bf16 *WlB[2], *WrB[2];
  for (int l = 0; l < 2; ++l){ WlB[l] = (bf16*)alloc(HID*HID*2); WrB[l] = (bf16*)alloc(HID*HID*2); }
  bf16* jkWb   = (bf16*)alloc(HID*256*2);
  bf16* pW1b   = (bf16*)alloc(HID*HID*2);
  bf16* pW2b   = (bf16*)alloc(64*HID*2);

  // ---- activations ----
  bf16*  xbf   = (bf16*)alloc((size_t)NND*TT*NFEAT*2);
  bf16*  seq1  = (bf16*)alloc((size_t)NND*TT*256*2);
  bf16*  seq2  = (bf16*)alloc((size_t)NND*TT*256*2);
  float* GF    = (float*)alloc((size_t)NND*512*4);
  float* GB    = (float*)alloc((size_t)NND*512*4);
  bf16*  hsF   = (bf16*)alloc((size_t)NND*HID*2);
  bf16*  hsB   = (bf16*)alloc((size_t)NND*HID*2);
  float* cF    = (float*)alloc((size_t)NND*HID*4);
  float* cB    = (float*)alloc((size_t)NND*HID*4);
  float* hF32  = (float*)alloc((size_t)NND*HID*4);
  bf16*  hBF   = (bf16*)alloc((size_t)NND*HID*2);
  float* sebuf = (float*)alloc((size_t)E*4);
  float* xlB   = (float*)alloc((size_t)NND*HID*4);
  float* xrB   = (float*)alloc((size_t)NND*HID*4);
  float* alpha = (float*)alloc((size_t)E*4*4);
  float* exb   = (float*)alloc((size_t)E*4*4);
  unsigned* amax = (unsigned*)alloc((size_t)NND*4*4);
  float* den   = (float*)alloc((size_t)NND*4*4);
  float* gout  = (float*)alloc((size_t)NND*HID*4);
  bf16*  jkbf  = (bf16*)alloc((size_t)NND*256*2);
  float* hcW   = (float*)alloc((size_t)NND*HID*4);
  float* meanB = (float*)alloc(HID*4);
  float* istdB = (float*)alloc(HID*4);
  bf16*  hfbf  = (bf16*)alloc((size_t)NND*HID*2);
  float* zp    = (float*)alloc((size_t)NND*HID*4);
  bf16*  zbf   = (bf16*)alloc((size_t)NND*HID*2);
  float* z2p   = (float*)alloc((size_t)NND*64*4);
  bf16*  z2bf  = (bf16*)alloc((size_t)NND*64*2);
  bf16*  ez1   = (bf16*)seq1;                        // alias: free after LSTM
  bf16*  ez2   = (bf16*)seq2;

  float* out_pred = (float*)d_out;
  float* out_hf   = out_pred + NND;
  float* out_fw   = out_hf + (size_t)NND*HID;
  float* out_se   = out_fw + NFEAT;

  auto cvt = [&](int idx, bf16* o, long n){
    tg_f2bf<<<dim3((unsigned)((n+255)/256)), dim3(256), 0, stream>>>((const float*)d_in[idx], o, n);
  };
  for (int d = 0; d < 4; ++d){
    cvt(lstm_base[d]+0, WihB[d], (long)512*lstm_ind[d]);
    cvt(lstm_base[d]+1, WhhB[d], (long)512*HID);
    tg_vadd<<<dim3(2), dim3(256), 0, stream>>>((const float*)d_in[lstm_base[d]+2],
                                               (const float*)d_in[lstm_base[d]+3], Bsum[d], 512);
  }
  cvt(19, projWb, HID*256); cvt(21, enW1b, HID*256); cvt(23, enW2b, 64*HID);
  cvt(27, WlB[0], HID*HID); cvt(29, WrB[0], HID*HID);
  cvt(34, WlB[1], HID*HID); cvt(36, WrB[1], HID*HID);
  cvt(45, jkWb, HID*256);   cvt(49, pW1b, HID*HID);  cvt(53, pW2b, 64*HID);

  // inp = x * feature_weights  -> bf16
  { long n = (long)NND*TT*NFEAT;
    tg_xcvt<<<dim3((unsigned)((n+255)/256)), dim3(256), 0, stream>>>(x, fw, xbf, n); }

  // ---- bi-LSTM, 2 layers ----
  const long gate_waves = (long)(NND/16) * (512/64);
  const int  cell_blk   = (NND*HID + 255)/256;
  for (int layer = 0; layer < 2; ++layer){
    const bf16* sin = layer ? seq1 : xbf;
    bf16* sout = layer ? seq2 : seq1;
    int in_d = layer ? 256 : NFEAT;
    int dF = layer*2, dB = layer*2 + 1;
    hipMemsetAsync(hsF, 0, (size_t)NND*HID*2, stream);
    hipMemsetAsync(hsB, 0, (size_t)NND*HID*2, stream);
    hipMemsetAsync(cF, 0, (size_t)NND*HID*4, stream);
    hipMemsetAsync(cB, 0, (size_t)NND*HID*4, stream);
    for (int s = 0; s < TT; ++s){
      int tf = s, tb = TT-1-s;
      tg_lstm_gate<<<dim3(gblk(gate_waves)), dim3(256), 0, stream>>>(sin, in_d, tf, hsF, WihB[dF], WhhB[dF], Bsum[dF], GF, NND);
      tg_lstm_gate<<<dim3(gblk(gate_waves)), dim3(256), 0, stream>>>(sin, in_d, tb, hsB, WihB[dB], WhhB[dB], Bsum[dB], GB, NND);
      tg_lstm_cell<<<dim3(cell_blk), dim3(256), 0, stream>>>(GF, cF, hsF, sout,       tf, NND);
      tg_lstm_cell<<<dim3(cell_blk), dim3(256), 0, stream>>>(GB, cB, hsB, sout + HID, tb, NND);
    }
  }

  // ---- proj: h = seq2[:, T-1, :] @ projW^T + b ----
  long wv128 = (long)(NND/16) * (HID/64);
  tg_gemm<<<dim3(gblk(wv128)), dim3(256), 0, stream>>>(seq2, (long)TT*256, (long)(TT-1)*256,
      projWb, 256, (const float*)d_in[20], hF32, hBF, NND, HID, 0, nullptr, nullptr);

  long ewv128 = (long)(E/16) * (HID/64);
  long ewv64  = (long)(E/16) * (64/64);
  auto edge_net = [&](float* se_out){
    tg_gemm<<<dim3(gblk(ewv128)), dim3(256), 0, stream>>>(hBF, 0, 0, enW1b, 256,
        (const float*)d_in[22], nullptr, ez1, E, HID, 1, src, dst);
    tg_gemm<<<dim3(gblk(ewv64)), dim3(256), 0, stream>>>(ez1, HID, 0, enW2b, HID,
        (const float*)d_in[24], nullptr, ez2, E, 64, 1, nullptr, nullptr);
    tg_dot<<<dim3((E+255)/256), dim3(256), 0, stream>>>(ez2, 64, (const float*)d_in[25],
        (const float*)d_in[26], se_out, E, 1);
  };
  edge_net(sebuf);

  // ---- 2 GATv2 layers ----
  for (int l = 0; l < 2; ++l){
    const int g0 = 27 + l*7;                 // Wl,bl,Wr,br,We,att,bias
    tg_gemm<<<dim3(gblk(wv128)), dim3(256), 0, stream>>>(hBF, HID, 0, WlB[l], HID,
        (const float*)d_in[g0+1], xlB, nullptr, NND, HID, 0, nullptr, nullptr);
    tg_gemm<<<dim3(gblk(wv128)), dim3(256), 0, stream>>>(hBF, HID, 0, WrB[l], HID,
        (const float*)d_in[g0+3], xrB, nullptr, NND, HID, 0, nullptr, nullptr);
    hipMemsetAsync(amax, 0, (size_t)NND*4*4, stream);
    hipMemsetAsync(den,  0, (size_t)NND*4*4, stream);
    hipMemsetAsync(gout, 0, (size_t)NND*HID*4, stream);
    tg_alpha<<<dim3((E+255)/256), dim3(256), 0, stream>>>(xlB, xrB, sebuf,
        (const float*)d_in[g0+4], (const float*)d_in[g0+5], src, dst, alpha, amax, E);
    tg_expden<<<dim3((E*4+255)/256), dim3(256), 0, stream>>>(alpha, amax, dst, exb, den, E);
    tg_scatter<<<dim3((E+255)/256), dim3(256), 0, stream>>>(exb, den, xlB, src, dst, gout, E);
    tg_ln<<<dim3(NND), dim3(HID), 0, stream>>>(hF32, gout, (const float*)d_in[g0+6],
        (const float*)d_in[41 + l*2], (const float*)d_in[42 + l*2], hF32, hBF, jkbf, l*HID);
    edge_net(l == 1 ? out_se : sebuf);       // final se goes straight to output
  }

  // ---- JK: h_final = relu(BN(jkbf @ jkW^T + b)) -> out_hf ----
  tg_gemm<<<dim3(gblk(wv128)), dim3(256), 0, stream>>>(jkbf, 256, 0, jkWb, 256,
      (const float*)d_in[46], hcW, nullptr, NND, HID, 0, nullptr, nullptr);
  tg_colstats<<<dim3(HID), dim3(256), 0, stream>>>(hcW, NND, HID, meanB, istdB);
  tg_bnrelu<<<dim3((NND*HID+255)/256), dim3(256), 0, stream>>>(hcW, meanB, istdB,
      (const float*)d_in[47], (const float*)d_in[48], out_hf, hfbf, (long)NND*HID, HID);

  // ---- predictor ----
  tg_gemm<<<dim3(gblk(wv128)), dim3(256), 0, stream>>>(hfbf, HID, 0, pW1b, HID,
      (const float*)d_in[50], zp, nullptr, NND, HID, 0, nullptr, nullptr);
  tg_colstats<<<dim3(HID), dim3(256), 0, stream>>>(zp, NND, HID, meanB, istdB);
  tg_bnrelu<<<dim3((NND*HID+255)/256), dim3(256), 0, stream>>>(zp, meanB, istdB,
      (const float*)d_in[51], (const float*)d_in[52], nullptr, zbf, (long)NND*HID, HID);
  long wv64 = (long)(NND/16) * (64/64);
  tg_gemm<<<dim3(gblk(wv64)), dim3(256), 0, stream>>>(zbf, HID, 0, pW2b, HID,
      (const float*)d_in[54], z2p, nullptr, NND, 64, 0, nullptr, nullptr);
  tg_colstats<<<dim3(64), dim3(256), 0, stream>>>(z2p, NND, 64, meanB, istdB);
  tg_bnrelu<<<dim3((NND*64+255)/256), dim3(256), 0, stream>>>(z2p, meanB, istdB,
      (const float*)d_in[55], (const float*)d_in[56], nullptr, z2bf, (long)NND*64, 64);
  tg_dot<<<dim3((NND+255)/256), dim3(256), 0, stream>>>(z2bf, 64, (const float*)d_in[57],
      (const float*)d_in[58], out_pred, NND, 0);

  // ---- feature_weights passthrough ----
  tg_copyf<<<dim3(1), dim3(64), 0, stream>>>(fw, out_fw, NFEAT);
}
